// LSTMLayer_27728308863512
// MI455X (gfx1250) — matrix operand order repaired and checked by reading, not compiled, not run on hardware
//
#include <hip/hip_runtime.h>
#include <hip/hip_bf16.h>

// ---------------------------------------------------------------------------
// LSTM (T=1024, B=128, D=256, H=256) + projection to C=128, for gfx1250.
//  Phase 1a: convert x to bf16 (grid-parallel, bandwidth bound).
//  Phase 1b: pack fp32 weights into bf16 WMMA B-fragment layout.
//  Phase 2 : persistent single-workgroup recurrence with
//            v_wmma_f32_16x16x32_bf16 + global_load_async_to_lds_b128 staging.
// ---------------------------------------------------------------------------

typedef __attribute__((ext_vector_type(16))) __bf16 v16bf;
typedef __attribute__((ext_vector_type(8)))  float  v8f;
typedef __attribute__((ext_vector_type(4)))  __bf16 bf16x4;

constexpr int T_STEPS = 1024;
constexpr int B       = 128;
constexpr int D       = 256;
constexpr int H       = 256;
constexpr int FOURH   = 4 * H;        // 1024
constexpr int KTOT    = D + H;        // 512
constexpr int NTILES  = FOURH / 16;   // 64 column tiles
constexpr int KTILES  = KTOT / 32;    // 16 K slices of 32
constexpr int NCLS    = 128;
constexpr int XSTEP   = B * D;        // 32768 elements per timestep

// ---------------------------------------------------------------------------
// Phase 1a: x fp32 -> bf16 (same [T][B][D] layout) into workspace.
// ---------------------------------------------------------------------------
__global__ void convert_x_kernel(const float* __restrict__ xf,
                                 __bf16* __restrict__ xb) {
    const size_t i = ((size_t)blockIdx.x * blockDim.x + threadIdx.x) * 4;
    const float4 v = *(const float4*)(xf + i);
    bf16x4 b;
    b.x = (__bf16)v.x; b.y = (__bf16)v.y;
    b.z = (__bf16)v.z; b.w = (__bf16)v.w;
    *(bf16x4*)(xb + i) = b;
}

// ---------------------------------------------------------------------------
// Phase 1b: pack fp32 kernel [KTOT, 4H] (row-major) into bf16 WMMA B-fragments.
// Block (nt, kt) -> 1024 bytes. Lane l holds 16 bf16 at byte offset l*32.
// 16-bit B 32x16 layout: lanes 0-15 = N, K 0..7,16..23; lanes 16-31 = K 8..15,24..31.
// ---------------------------------------------------------------------------
__global__ void pack_w_kernel(const float* __restrict__ Wf, __bf16* __restrict__ Wp) {
    const int blk = blockIdx.x;            // nt*KTILES + kt
    const int nt  = blk / KTILES;
    const int kt  = blk % KTILES;
    const int l   = threadIdx.x;           // 0..31
    const int n   = nt * 16 + (l & 15);
    const int kb  = kt * 32 + ((l >> 4) ? 8 : 0);
    __bf16* dst = Wp + (size_t)blk * 512 + l * 16;
#pragma unroll
    for (int e = 0; e < 16; ++e) {
        dst[e] = (__bf16)Wf[(size_t)(kb + e + (e >= 8 ? 8 : 0)) * FOURH + n];
    }
}

// fast activations: v_exp_f32 + raw v_rcp_f32 (no IEEE div fixup sequences)
__device__ __forceinline__ float sigmoidf_(float x) {
    return __builtin_amdgcn_rcpf(1.0f + __expf(-x));
}
__device__ __forceinline__ float tanhf_(float x) {
    return 1.0f - 2.0f * __builtin_amdgcn_rcpf(1.0f + __expf(2.0f * x));
}

// Stage one 64KB bf16 x-tile into LDS with async copies (ASYNCcnt path).
// Per thread: 8 x 16B transfers. IOFFSET is added to BOTH the global and the
// LDS address, and our global/LDS tile layouts are identical, so one base
// address + immediate offsets covers the tile.
__device__ __forceinline__ void stage_x_async(const __bf16* gsrc,
                                              unsigned lds_byte) {
    const unsigned long long ga = (unsigned long long)gsrc;
#pragma unroll
    for (int i = 0; i < 8; ++i) {
        asm volatile("global_load_async_to_lds_b128 %0, %1, off offset:%c2"
                     :: "v"(lds_byte), "v"(ga), "i"(i * 8192)
                     : "memory");
    }
}
__device__ __forceinline__ void wait_async_done() {
    asm volatile("s_wait_asynccnt 0" ::: "memory");
}

union AFrag { v16bf v; uint4 q[2]; };
union CFrag { v8f  v; float f[8]; };

// one K-slice: load A fragment from LDS, 4 gate WMMAs from packed weights
__device__ __forceinline__ void kslice(const __bf16* __restrict__ asrc, int grp,
                                       const __bf16* __restrict__ wp_base,
                                       CFrag acc[4]) {
    AFrag a;
    a.q[0] = *(const uint4*)(asrc + grp * 8);       // K 0..7   (or 8..15)
    a.q[1] = *(const uint4*)(asrc + 16 + grp * 8);  // K 16..23 (or 24..31)
#pragma unroll
    for (int g = 0; g < 4; ++g) {
        const __bf16* bsrc = wp_base + (size_t)g * 16 * KTILES * 512;
        AFrag b;
        b.q[0] = *(const uint4*)(bsrc);
        b.q[1] = *(const uint4*)(bsrc + 8);
        acc[g].v = __builtin_amdgcn_wmma_f32_16x16x32_bf16(
            false, a.v, false, b.v, (short)0, acc[g].v, false, false);
    }
}

// ---------------------------------------------------------------------------
// Phase 2: persistent single-workgroup LSTM + final projection.
//   LDS: xbuf[2][B][D] bf16 (128KB) + hbuf[2][B][H] bf16 (128KB) = 256KB.
//   16 waves: wave = bt(0..7) + 8*half(0..1). Each wave: 8 h-tiles x 4 gates.
//   Cell state c lives in VGPRs. One barrier per timestep (double buffers).
// ---------------------------------------------------------------------------
__launch_bounds__(512, 1)
__global__ void lstm_fused_kernel(const __bf16* __restrict__ xb,
                                  const float* __restrict__ bias,
                                  const __bf16* __restrict__ Wp,
                                  const float* __restrict__ wproj,
                                  const float* __restrict__ bproj,
                                  float* __restrict__ out) {
    extern __shared__ __bf16 smem[];
    __bf16* xbuf = smem;                 // 2 * B * D
    __bf16* hbuf = smem + 2 * B * D;     // 2 * B * H

    const int tid  = threadIdx.x;
    const int lane = tid & 31;
    const int wave = tid >> 5;
    const int bt   = wave & 7;           // batch tile (16 rows)
    const int half = wave >> 3;          // which 8 h-tiles
    const int rit  = lane & 15;          // row-in-tile (A) / col (B,C)
    const int grp  = lane >> 4;
    const int arow = bt * 16 + rit;      // A-matrix row for this lane

    // init h_{-1} = 0 (buffer 0)
    for (int i = tid; i < B * H; i += 512) hbuf[i] = (__bf16)0.0f;

    // preload per-lane gate biases: col = g*H + (half*8+j)*16 + rit
    // forget-gate bias (+1.0, TF BasicLSTMCell) folded in here once.
    float breg[4][8];
#pragma unroll
    for (int g = 0; g < 4; ++g)
#pragma unroll
        for (int j = 0; j < 8; ++j)
            breg[g][j] = bias[g * H + (half * 8 + j) * 16 + rit]
                         + (g == 2 ? 1.0f : 0.0f);

    // cell state in VGPRs: c[j][r] <-> row bt*16+grp*8+r, col (half*8+j)*16+rit
    float creg[8][8];
#pragma unroll
    for (int j = 0; j < 8; ++j)
#pragma unroll
        for (int r = 0; r < 8; ++r) creg[j][r] = 0.0f;

    // stage x_0 into xbuf[0] asynchronously
    stage_x_async(xb + (size_t)tid * 8, (unsigned)(tid * 16));
    wait_async_done();
    __syncthreads();

    for (int t = 0; t < T_STEPS; ++t) {
        const __bf16* xcur = xbuf + (t & 1) * XSTEP;
        const __bf16* hcur = hbuf + (t & 1) * (B * H);
        __bf16*       hnxt = hbuf + ((t + 1) & 1) * (B * H);

        // async prefetch of x_{t+1} into the other x buffer (ASYNCcnt)
        if (t + 1 < T_STEPS) {
            stage_x_async(xb + (size_t)(t + 1) * XSTEP + (size_t)tid * 8,
                          (unsigned)((((t + 1) & 1) * XSTEP + tid * 8) * 2));
        }

#pragma unroll
        for (int j = 0; j < 8; ++j) {
            // seed accumulators with the gate bias (uniform across the 8 rows
            // a lane holds, since bias depends only on the column)
            CFrag acc[4];
#pragma unroll
            for (int g = 0; g < 4; ++g)
#pragma unroll
                for (int r = 0; r < 8; ++r) acc[g].f[r] = breg[g][j];

            // base of packed W for gate 0, this h-tile: nt = half*8 + j
            const __bf16* wpj =
                Wp + ((size_t)(half * 8 + j) * KTILES) * 512 + lane * 16;

            // K-slices 0..7: A rows come from x_t
#pragma unroll 2
            for (int kb = 0; kb < 8; ++kb)
                kslice(xcur + arow * D + kb * 32, grp,
                       wpj + (size_t)kb * 512, acc);
            // K-slices 8..15: A rows come from h_{t-1}
#pragma unroll 2
            for (int kb = 8; kb < 16; ++kb)
                kslice(hcur + arow * H + (kb - 8) * 32, grp,
                       wpj + (size_t)kb * 512, acc);

            // gate fusion: TF order i, j, f, o
            const int col = (half * 8 + j) * 16 + rit;
#pragma unroll
            for (int r = 0; r < 8; ++r) {
                const float iv = sigmoidf_(acc[0].f[r]);
                const float jv = tanhf_  (acc[1].f[r]);
                const float fv = sigmoidf_(acc[2].f[r]);   // bias+1 pre-folded
                const float ov = sigmoidf_(acc[3].f[r]);
                const float cn = creg[j][r] * fv + iv * jv;
                creg[j][r] = cn;
                hnxt[(bt * 16 + grp * 8 + r) * H + col] =
                    (__bf16)(tanhf_(cn) * ov);
            }
        }
        wait_async_done();   // this wave's x_{t+1} tile is in LDS
        __syncthreads();     // everyone's writes visible before next step
    }

    // ------------------------------------------------------------------
    // Final projection: out[128][128] = h_last @ wproj + bproj  (fp32)
    // h_last lives in hbuf buffer (T_STEPS & 1) == 0.
    // ------------------------------------------------------------------
    const __bf16* hfin = hbuf;           // buffer 0
    const int row = tid >> 2;            // 0..127
    const int c0  = (tid & 3) * 32;      // 32 cols per thread
    float pacc[32];
#pragma unroll
    for (int cc = 0; cc < 32; ++cc) pacc[cc] = 0.0f;

    for (int k = 0; k < H; ++k) {
        const float hv = (float)hfin[row * H + k];
        const float* wr = wproj + (size_t)k * NCLS + c0;
#pragma unroll
        for (int cq = 0; cq < 8; ++cq) {
            const float4 wv = *(const float4*)(wr + cq * 4);
            pacc[cq * 4 + 0] += hv * wv.x;
            pacc[cq * 4 + 1] += hv * wv.y;
            pacc[cq * 4 + 2] += hv * wv.z;
            pacc[cq * 4 + 3] += hv * wv.w;
        }
    }
#pragma unroll
    for (int cc = 0; cc < 32; ++cc)
        out[row * NCLS + c0 + cc] = pacc[cc] + bproj[c0 + cc];
}

// ---------------------------------------------------------------------------
extern "C" void kernel_launch(void* const* d_in, const int* in_sizes, int n_in,
                              void* d_out, int out_size, void* d_ws, size_t ws_size,
                              hipStream_t stream) {
    (void)in_sizes; (void)n_in; (void)out_size; (void)ws_size;
    const float* x      = (const float*)d_in[0];   // [T, B, D]
    const float* kernel = (const float*)d_in[1];   // [D+H, 4H]
    const float* bias   = (const float*)d_in[2];   // [4H]
    const float* wproj  = (const float*)d_in[3];   // [H, C]
    const float* bproj  = (const float*)d_in[4];   // [C]
    float* out = (float*)d_out;                    // [B, C]

    __bf16* Wp  = (__bf16*)d_ws;                   // 512*1024 bf16 = 1 MB
    __bf16* xbf = Wp + (size_t)KTOT * FOURH;       // [T][B][D] bf16 = 64 MB

    // Phase 1 (grid-parallel, whole GPU)
    convert_x_kernel<<<(T_STEPS * XSTEP) / (256 * 4), 256, 0, stream>>>(x, xbf);
    pack_w_kernel<<<NTILES * KTILES, 32, 0, stream>>>(kernel, Wp);

    // Phase 2 (persistent workgroup)
    const size_t lds_bytes = (size_t)(2 * B * D + 2 * B * H) * sizeof(__bf16); // 256 KB
    lstm_fused_kernel<<<1, 512, lds_bytes, stream>>>(xbf, bias, Wp, wproj, bproj, out);
}
